// MulLayer_19421842113038
// MI455X (gfx1250) — matrix operand order repaired and checked by reading, not compiled, not run on hardware
//
#include <hip/hip_runtime.h>
#include <hip/hip_bf16.h>

// ---------------------------------------------------------------------------
// MulLayer (style-transfer "mul layer") for MI455X / gfx1250.
//  - 3x3 / 1x1 convs: implicit GEMM on v_wmma_f32_16x16x32_bf16 (fp32 accum),
//    weights streamed via CDNA5 async DMA (global_load_async_to_lds_b128 +
//    s_wait_asynccnt), activations staged sync with zero-pad halo.
//  - Gram matrices: v_wmma_f32_16x16x4_f32 (full fp32, matches reference).
//  - Small ops (means, VAE, fc, einsums) scalar fp32.
// ---------------------------------------------------------------------------

typedef __attribute__((ext_vector_type(16))) __bf16 v16bf;
typedef __attribute__((ext_vector_type(8)))  float  v8f;
typedef __attribute__((ext_vector_type(2)))  float  v2f;

struct Frag32 { uint4 lo, hi; };   // 32 bytes == v16bf

__device__ __forceinline__ unsigned short f2bf(float f) {
  union { float f; unsigned u; } c; c.f = f;
  unsigned u = c.u;
  unsigned r = (u + 0x7FFFu + ((u >> 16) & 1u)) >> 16;   // round-nearest-even
  return (unsigned short)r;
}

// ------------------------- spatial means (4096 per row) ---------------------
__global__ __launch_bounds__(256)
void mean_kernel(const float* __restrict__ cF, const float* __restrict__ sF1,
                 const float* __restrict__ sF2,
                 float* __restrict__ cM, float* __restrict__ sM1,
                 float* __restrict__ sM2) {
  __shared__ float red[256];
  const int row = blockIdx.x;                 // b*512 + c, 0..2047
  const int which = blockIdx.y;               // 0:cF 1:sF1 2:sF2
  const float* src = (which == 0) ? cF : (which == 1) ? sF1 : sF2;
  float* dst = (which == 0) ? cM : (which == 1) ? sM1 : sM2;
  const float* p = src + (long)row * 4096;
  float s = 0.f;
  for (int i = threadIdx.x; i < 4096; i += 256) s += p[i];
  red[threadIdx.x] = s;
  __syncthreads();
  for (int t = 128; t > 0; t >>= 1) {
    if (threadIdx.x < t) red[threadIdx.x] += red[threadIdx.x + t];
    __syncthreads();
  }
  if (threadIdx.x == 0) dst[row] = red[0] * (1.f / 4096.f);
}

// ------------------------- VAE (single block, LDS-resident) -----------------
__global__ __launch_bounds__(256)
void vae_kernel(const float* __restrict__ sm1, const float* __restrict__ sm2,
                const float* __restrict__ noise_a, const float* __restrict__ noise_b,
                const float* __restrict__ kp,
                const float* __restrict__ enc_w, const float* __restrict__ enc_b,
                const float* __restrict__ bn_g, const float* __restrict__ bn_b,
                const float* __restrict__ d1_w, const float* __restrict__ d1_b,
                const float* __restrict__ bn2_g, const float* __restrict__ bn2_b,
                const float* __restrict__ d2_w, const float* __restrict__ d2_b,
                float* __restrict__ smean_out, float* __restrict__ kl_out) {
  __shared__ float X[8][512];
  __shared__ float E[8][256];
  __shared__ float ZA[4][128];
  __shared__ float ZB[4][128];
  __shared__ float Z[4][128];
  __shared__ float D[4][256];
  __shared__ float red[256];
  const int tid = threadIdx.x;
  const float k = kp[0];

  for (int i = tid; i < 4096; i += 256) {
    int r = i >> 9, c = i & 511;
    X[r][c] = (r < 4) ? sm1[r * 512 + c] : sm2[(r - 4) * 512 + c];
  }
  __syncthreads();
  for (int o = tid; o < 2048; o += 256) {          // enc = X @ enc_w^T + enc_b
    int r = o >> 8, c = o & 255;
    float s = enc_b[c];
    const float* w = enc_w + (long)c * 512;
    for (int i = 0; i < 512; ++i) s += X[r][i] * w[i];
    E[r][c] = s;
  }
  __syncthreads();
  float klp = 0.f;
  {
    const int c = tid & 127;
    const int grp = tid >> 7;
    const int rb = grp * 4;
    float m0 = E[rb + 0][c], m1 = E[rb + 1][c], m2 = E[rb + 2][c], m3 = E[rb + 3][c];
    float mu = 0.25f * (m0 + m1 + m2 + m3);
    float var = 0.25f * ((m0 - mu) * (m0 - mu) + (m1 - mu) * (m1 - mu) +
                         (m2 - mu) * (m2 - mu) + (m3 - mu) * (m3 - mu));
    float rs = rsqrtf(var + 1e-5f);
    const float* nz = grp ? noise_b : noise_a;
    const float wk = grp ? (1.f - k) : k;
    for (int r = 0; r < 4; ++r) {
      float mval = E[rb + r][c];
      float ls = E[rb + r][128 + c];
      float el = expf(ls);
      float bn = (mval - mu) * rs * bn_g[c] + bn_b[c];
      float samp = bn + el * nz[r * 128 + c];
      if (grp) ZB[r][c] = samp; else ZA[r][c] = samp;
      klp += wk * (0.5f * (mval * mval + el * el - 1.f) - ls);
    }
  }
  red[tid] = klp;
  __syncthreads();
  for (int t = 128; t > 0; t >>= 1) {
    if (tid < t) red[tid] += red[tid + t];
    __syncthreads();
  }
  if (tid == 0) kl_out[0] = red[0];
  for (int i = tid; i < 512; i += 256) {
    int r = i >> 7, c = i & 127;
    Z[r][c] = k * ZA[r][c] + (1.f - k) * ZB[r][c];
  }
  __syncthreads();
  for (int o = tid; o < 1024; o += 256) {          // d1
    int r = o >> 8, c = o & 255;
    float s = d1_b[c];
    const float* w = d1_w + (long)c * 128;
    for (int i = 0; i < 128; ++i) s += Z[r][i] * w[i];
    D[r][c] = s;
  }
  __syncthreads();
  {
    const int c = tid;                              // bn2 + relu
    float m0 = D[0][c], m1 = D[1][c], m2 = D[2][c], m3 = D[3][c];
    float mu = 0.25f * (m0 + m1 + m2 + m3);
    float var = 0.25f * ((m0 - mu) * (m0 - mu) + (m1 - mu) * (m1 - mu) +
                         (m2 - mu) * (m2 - mu) + (m3 - mu) * (m3 - mu));
    float rs = rsqrtf(var + 1e-5f);
    for (int r = 0; r < 4; ++r) {
      float v = (D[r][c] - mu) * rs * bn2_g[c] + bn2_b[c];
      D[r][c] = v > 0.f ? v : 0.f;
    }
  }
  __syncthreads();
  for (int o = tid; o < 2048; o += 256) {          // d2 -> smean_out
    int r = o >> 9, c = o & 511;
    float s = d2_b[c];
    const float* w = d2_w + (long)c * 256;
    for (int i = 0; i < 256; ++i) s += D[r][i] * w[i];
    smean_out[r * 512 + c] = s;
  }
}

// ------------------------- bf16 activation prep -----------------------------
__global__ __launch_bounds__(256)
void prep_c_kernel(const float* __restrict__ cF, const float* __restrict__ cM,
                   unsigned short* __restrict__ dst) {
  long idx = (long)blockIdx.x * 256 + threadIdx.x;
  int row = (int)(idx >> 12);
  dst[idx] = f2bf(cF[idx] - cM[row]);
}

__global__ __launch_bounds__(256)
void prep_s_kernel(const float* __restrict__ sF1, const float* __restrict__ sF2,
                   const float* __restrict__ sMean, const float* __restrict__ kp,
                   unsigned short* __restrict__ dst) {
  long idx = (long)blockIdx.x * 256 + threadIdx.x;
  int row = (int)(idx >> 12);
  float k = kp[0];
  dst[idx] = f2bf(k * sF1[idx] + (1.f - k) * sF2[idx] - sMean[row]);
}

// weights [Cout][Cin][RS] fp32 -> [RS][Cout][Cin] bf16 (rows contiguous in ci
// so the weight tile can be DMA'd to LDS without transpose)
__global__ __launch_bounds__(256)
void wprep_kernel(const float* __restrict__ w, unsigned short* __restrict__ wt,
                  int Cin, int Cout, int RS) {
  int idx = blockIdx.x * 256 + threadIdx.x;
  int total = RS * Cin * Cout;
  if (idx >= total) return;
  int pos = idx / (Cin * Cout);
  int rem = idx - pos * Cin * Cout;
  int co = rem / Cin;
  int ci = rem - co * Cin;
  wt[idx] = f2bf(w[((long)co * Cin + ci) * RS + pos]);
}

// ------------------------- implicit-GEMM conv (WMMA bf16) -------------------
// GEMM roles: A = activations (M = spatial), B = weights (N = Cout), K = Cin.
// in  : [B][Cin][4096] bf16      wt : [RS][Cout][Cin] bf16
// Block = 256 thr = 8 waves; block tile = 128 spatial (2 rows) x 32 co,
// K-chunk = 64 -> 4 WMMAs per wave per stage (2 K-sub x 2 co-frags).
// Weights -> LDS via async DMA (contiguous, no zero-fill needed);
// activations -> LDS sync (transpose [ci][x] -> [x][ci] + zero-pad halo).
template <int RS, bool RELU, bool OUT_BF16>
__global__ __launch_bounds__(256)
void conv_wmma(const unsigned short* __restrict__ in,
               const unsigned short* __restrict__ wt,
               const float* __restrict__ bias,
               void* __restrict__ outv, int Cin, int Cout) {
  __shared__ __align__(16) unsigned short Xs[128][72];  // [spatial][k] (+pad)
  __shared__ __align__(16) unsigned short Ws[32][72];   // [co][k]      (+pad)

  const int tid = threadIdx.x;
  const int lane = tid & 31;
  const int wv = tid >> 5;        // 0..7, wave's 16-row spatial tile
  const int half = lane >> 4;
  const int l16 = lane & 15;

  const int spB = blockIdx.x * 128;     // spatial base (two image rows)
  const int y0 = spB >> 6;
  const int coB = blockIdx.y * 32;
  const int b = blockIdx.z;

  v8f acc0 = {}, acc1 = {};
  const unsigned short* inB = in + (long)b * Cin * 4096;

  for (int pos = 0; pos < RS; ++pos) {
    const int dy = (RS == 9) ? (pos / 3) : 0;
    const int dx = (RS == 9) ? (pos % 3) : 0;
    const unsigned short* wp = wt + (long)pos * Cout * Cin + (long)coB * Cin;
    for (int ci0 = 0; ci0 < Cin; ci0 += 64) {
      __syncthreads();
      // weights tile (32 co x 64 k = 4KB) via CDNA5 async DMA:
      // 256 lanes x 16B, per-lane (lds, global) address pairs.
      {
        const int co = tid >> 3;
        const int ch = tid & 7;                       // 16B chunk
        const unsigned short* g = wp + (long)co * Cin + ci0 + ch * 8;
        unsigned ldsoff = (unsigned)(size_t)&Ws[co][ch * 8];
        asm volatile("global_load_async_to_lds_b128 %0, %1, off"
                     :: "v"(ldsoff), "v"(g) : "memory");
      }
      // activation tile (128 sp x 64 k = 16KB), zero-padded halo
      for (int t = 0; t < 32; ++t) {
        int idx = t * 256 + tid;
        int k = idx >> 7;                             // 0..63
        int n = idx & 127;                            // spatial in tile
        int iy = y0 + (n >> 6) + ((RS == 9) ? (dy - 1) : 0);
        int ix = (n & 63) + ((RS == 9) ? (dx - 1) : 0);
        unsigned short v = 0;
        if ((unsigned)iy < 64u && (unsigned)ix < 64u)
          v = inB[(long)(ci0 + k) * 4096 + iy * 64 + ix];
        Xs[n][k] = v;
      }
      asm volatile("s_wait_asynccnt 0x0" ::: "memory");
      __syncthreads();
      // 2 K-subchunks x 2 co-fragments = 4 WMMAs, A-fragment reused
      const unsigned short* arow = &Xs[wv * 16 + l16][0];
      const unsigned short* b0 = &Ws[l16][0];
      const unsigned short* b1 = &Ws[16 + l16][0];
      #pragma unroll
      for (int s = 0; s < 2; ++s) {
        Frag32 fa, fb0, fb1;
        fa.lo = *(const uint4*)(arow + s * 32 + half * 16);
        fa.hi = *(const uint4*)(arow + s * 32 + half * 16 + 8);
        fb0.lo = *(const uint4*)(b0 + s * 32 + half * 16);
        fb0.hi = *(const uint4*)(b0 + s * 32 + half * 16 + 8);
        fb1.lo = *(const uint4*)(b1 + s * 32 + half * 16);
        fb1.hi = *(const uint4*)(b1 + s * 32 + half * 16 + 8);
        v16bf av = __builtin_bit_cast(v16bf, fa);
        acc0 = __builtin_amdgcn_wmma_f32_16x16x32_bf16(
            false, av, false, __builtin_bit_cast(v16bf, fb0), (short)0, acc0,
            false, false);
        acc1 = __builtin_amdgcn_wmma_f32_16x16x32_bf16(
            false, av, false, __builtin_bit_cast(v16bf, fb1), (short)0, acc1,
            false, false);
      }
    }
  }
  // D layout: M(spatial) = r + 8*half, N(co) = l16
  #pragma unroll
  for (int cb = 0; cb < 2; ++cb) {
    const int co = coB + cb * 16 + l16;
    const float bsv = bias[co];
    v8f a = cb ? acc1 : acc0;
    #pragma unroll
    for (int r = 0; r < 8; ++r) {
      int sp = spB + wv * 16 + half * 8 + r;
      float v = a[r] + bsv;
      if (RELU) v = v > 0.f ? v : 0.f;
      long oidx = ((long)b * Cout + co) * 4096 + sp;
      if (OUT_BF16) ((unsigned short*)outv)[oidx] = f2bf(v);
      else          ((float*)outv)[oidx] = v;
    }
  }
}

// ------------------------- gram via v_wmma_f32_16x16x4_f32 ------------------
// gram[b] = h[b] (32x4096) @ h[b]^T / 4096, fp32.  One wave per image:
// 2x2 fragment grid; A and B fragments per lane are identical loads.
__global__ __launch_bounds__(32)
void gram_wmma_kernel(const float* __restrict__ h, float* __restrict__ gram) {
  const int b = blockIdx.x;
  const float* hb = h + (long)b * 32 * 4096;
  const int lane = threadIdx.x;
  const int half = lane >> 4;
  const int l16 = lane & 15;
  v8f a00 = {}, a01 = {}, a10 = {}, a11 = {};
  const float* r0 = hb + (long)l16 * 4096;          // rows 0..15
  const float* r1 = hb + (long)(16 + l16) * 4096;   // rows 16..31
  for (int k0 = 0; k0 < 4096; k0 += 4) {
    const int k = k0 + 2 * half;                    // 16x4 f32 A: K = 2*half+e
    v2f f0 = { r0[k], r0[k + 1] };
    v2f f1 = { r1[k], r1[k + 1] };
    a00 = __builtin_amdgcn_wmma_f32_16x16x4_f32(false, f0, false, f0, (short)0, a00, false, false);
    a01 = __builtin_amdgcn_wmma_f32_16x16x4_f32(false, f0, false, f1, (short)0, a01, false, false);
    a10 = __builtin_amdgcn_wmma_f32_16x16x4_f32(false, f1, false, f0, (short)0, a10, false, false);
    a11 = __builtin_amdgcn_wmma_f32_16x16x4_f32(false, f1, false, f1, (short)0, a11, false, false);
  }
  #pragma unroll
  for (int r = 0; r < 8; ++r) {
    int mi = half * 8 + r;
    gram[b * 1024 + (mi) * 32 + l16]           = a00[r] * (1.f / 4096.f);
    gram[b * 1024 + (mi) * 32 + 16 + l16]      = a01[r] * (1.f / 4096.f);
    gram[b * 1024 + (16 + mi) * 32 + l16]      = a10[r] * (1.f / 4096.f);
    gram[b * 1024 + (16 + mi) * 32 + 16 + l16] = a11[r] * (1.f / 4096.f);
  }
}

// ------------------------- fc / einsums -------------------------------------
__global__ __launch_bounds__(256)
void fc_kernel(const float* __restrict__ g, const float* __restrict__ fcw,
               const float* __restrict__ fcb, float* __restrict__ mat) {
  int o = blockIdx.x * 256 + threadIdx.x;   // 0..4095
  int b = o >> 10, c = o & 1023;
  const float* x = g + b * 1024;
  const float* w = fcw + (long)c * 1024;
  float s = fcb[c];
  for (int i = 0; i < 1024; ++i) s += x[i] * w[i];
  mat[o] = s;
}

__global__ __launch_bounds__(1024)
void trans_kernel(const float* __restrict__ matS, const float* __restrict__ matC,
                  float* __restrict__ trans) {
  int b = blockIdx.x, t = threadIdx.x;
  int i = t >> 5, kk = t & 31;
  const float* S = matS + b * 1024;
  const float* C = matC + b * 1024;
  float s = 0.f;
  #pragma unroll
  for (int j = 0; j < 32; ++j) s += S[i * 32 + j] * C[j * 32 + kk];
  trans[b * 1024 + t] = s;
}

__global__ __launch_bounds__(256)
void m2_kernel(const float* __restrict__ uzw, const float* __restrict__ trans,
               float* __restrict__ m2) {
  int idx = blockIdx.x * 256 + threadIdx.x;   // 4*512*32
  int b = idx >> 14;
  int rem = idx & 16383;
  int c = rem >> 5, j = rem & 31;
  float s = 0.f;
  #pragma unroll
  for (int m = 0; m < 32; ++m) s += uzw[c * 32 + m] * trans[b * 1024 + m * 32 + j];
  m2[idx] = s;
}

__global__ __launch_bounds__(256)
void final_kernel(const float* __restrict__ m2, const float* __restrict__ cc,
                  const float* __restrict__ uzb, const float* __restrict__ smean,
                  float* __restrict__ out) {
  long idx = (long)blockIdx.x * 256 + threadIdx.x;   // 8388608
  int sp = (int)(idx & 4095);
  int bc = (int)(idx >> 12);
  int b = bc >> 9, c = bc & 511;
  const float* mrow = m2 + ((long)bc << 5);
  const float* ccb = cc + (long)b * 32 * 4096 + sp;
  float s = uzb[c] + smean[bc];
  #pragma unroll
  for (int j = 0; j < 32; ++j) s += mrow[j] * ccb[(long)j * 4096];
  out[idx] = s;
}

// ---------------------------------------------------------------------------
extern "C" void kernel_launch(void* const* d_in, const int* in_sizes, int n_in,
                              void* d_out, int out_size, void* d_ws, size_t ws_size,
                              hipStream_t stream) {
  (void)in_sizes; (void)n_in; (void)out_size; (void)ws_size;
  // d_in order = jax tree-flatten (alphabetical dict keys per level), see r0.
  const float* cF      = (const float*)d_in[0];
  const float* kp      = (const float*)d_in[1];
  const float* noise_a = (const float*)d_in[2];
  const float* noise_b = (const float*)d_in[3];
  const float* cnet_b1 = (const float*)d_in[4];
  const float* cnet_b2 = (const float*)d_in[5];
  const float* cnet_b3 = (const float*)d_in[6];
  const float* cnet_fcb= (const float*)d_in[7];
  const float* cnet_fcw= (const float*)d_in[8];
  const float* cnet_w1 = (const float*)d_in[9];
  const float* cnet_w2 = (const float*)d_in[10];
  const float* cnet_w3 = (const float*)d_in[11];
  const float* comp_b  = (const float*)d_in[12];
  const float* comp_w  = (const float*)d_in[13];
  const float* snet_b1 = (const float*)d_in[14];
  const float* snet_b2 = (const float*)d_in[15];
  const float* snet_b3 = (const float*)d_in[16];
  const float* snet_fcb= (const float*)d_in[17];
  const float* snet_fcw= (const float*)d_in[18];
  const float* snet_w1 = (const float*)d_in[19];
  const float* snet_w2 = (const float*)d_in[20];
  const float* snet_w3 = (const float*)d_in[21];
  const float* uzb     = (const float*)d_in[22];
  const float* uzw     = (const float*)d_in[23];
  const float* bn2_b   = (const float*)d_in[24];
  const float* bn2_g   = (const float*)d_in[25];
  const float* bn_b    = (const float*)d_in[26];
  const float* bn_g    = (const float*)d_in[27];
  const float* d1_b    = (const float*)d_in[28];
  const float* d1_w    = (const float*)d_in[29];
  const float* d2_b    = (const float*)d_in[30];
  const float* d2_w    = (const float*)d_in[31];
  const float* enc_b   = (const float*)d_in[32];
  const float* enc_w   = (const float*)d_in[33];
  const float* sF1     = (const float*)d_in[34];
  const float* sF2     = (const float*)d_in[35];

  float* out       = (float*)d_out;            // [4,512,64,64]
  float* trans_out = out + 8388608;            // [4,32,32]
  float* kl_out    = out + 8392704;            // scalar

  char* wsp = (char*)d_ws;
  auto alloc = [&](size_t bytes) -> void* {
    void* p = wsp;
    wsp += (bytes + 255) & ~(size_t)255;
    return p;
  };
  float* cM    = (float*)alloc(2048 * 4);
  float* sM1   = (float*)alloc(2048 * 4);
  float* sM2   = (float*)alloc(2048 * 4);
  float* sMout = (float*)alloc(2048 * 4);
  unsigned short* inc = (unsigned short*)alloc((size_t)8388608 * 2);
  unsigned short* ins = (unsigned short*)alloc((size_t)8388608 * 2);
  unsigned short* w1t = (unsigned short*)alloc((size_t)9 * 512 * 256 * 2);
  unsigned short* w2t = (unsigned short*)alloc((size_t)9 * 256 * 128 * 2);
  unsigned short* w3t = (unsigned short*)alloc((size_t)9 * 128 * 32 * 2);
  unsigned short* cwt = (unsigned short*)alloc((size_t)512 * 32 * 2);
  unsigned short* h1  = (unsigned short*)alloc((size_t)4 * 256 * 4096 * 2);
  unsigned short* h2  = (unsigned short*)alloc((size_t)4 * 128 * 4096 * 2);
  float* h3     = (float*)alloc((size_t)4 * 32 * 4096 * 4);
  float* gramB  = (float*)alloc(4096 * 4);
  float* matC   = (float*)alloc(4096 * 4);
  float* matS   = (float*)alloc(4096 * 4);
  float* ccflat = (float*)alloc((size_t)4 * 32 * 4096 * 4);
  float* m2buf  = (float*)alloc((size_t)4 * 512 * 32 * 4);

  mean_kernel<<<dim3(2048, 3), 256, 0, stream>>>(cF, sF1, sF2, cM, sM1, sM2);
  vae_kernel<<<1, 256, 0, stream>>>(sM1, sM2, noise_a, noise_b, kp,
                                    enc_w, enc_b, bn_g, bn_b, d1_w, d1_b,
                                    bn2_g, bn2_b, d2_w, d2_b, sMout, kl_out);
  prep_c_kernel<<<32768, 256, 0, stream>>>(cF, cM, inc);
  prep_s_kernel<<<32768, 256, 0, stream>>>(sF1, sF2, sMout, kp, ins);

  // compress (1x1 conv) -> cc_flat [4,32,4096] fp32
  wprep_kernel<<<(512 * 32 + 255) / 256, 256, 0, stream>>>(comp_w, cwt, 512, 32, 1);
  conv_wmma<1, false, false><<<dim3(32, 1, 4), 256, 0, stream>>>(inc, cwt, comp_b,
                                                                 ccflat, 512, 32);
  // cnet: 512->256->128->32, gram, fc -> matC
  wprep_kernel<<<9 * 512 * 256 / 256, 256, 0, stream>>>(cnet_w1, w1t, 512, 256, 9);
  conv_wmma<9, true, true><<<dim3(32, 8, 4), 256, 0, stream>>>(inc, w1t, cnet_b1, h1, 512, 256);
  wprep_kernel<<<9 * 256 * 128 / 256, 256, 0, stream>>>(cnet_w2, w2t, 256, 128, 9);
  conv_wmma<9, true, true><<<dim3(32, 4, 4), 256, 0, stream>>>(h1, w2t, cnet_b2, h2, 256, 128);
  wprep_kernel<<<9 * 128 * 32 / 256, 256, 0, stream>>>(cnet_w3, w3t, 128, 32, 9);
  conv_wmma<9, false, false><<<dim3(32, 1, 4), 256, 0, stream>>>(h2, w3t, cnet_b3, h3, 128, 32);
  gram_wmma_kernel<<<4, 32, 0, stream>>>(h3, gramB);
  fc_kernel<<<16, 256, 0, stream>>>(gramB, cnet_fcw, cnet_fcb, matC);

  // snet (same buffers, sequential on stream) -> matS
  wprep_kernel<<<9 * 512 * 256 / 256, 256, 0, stream>>>(snet_w1, w1t, 512, 256, 9);
  conv_wmma<9, true, true><<<dim3(32, 8, 4), 256, 0, stream>>>(ins, w1t, snet_b1, h1, 512, 256);
  wprep_kernel<<<9 * 256 * 128 / 256, 256, 0, stream>>>(snet_w2, w2t, 256, 128, 9);
  conv_wmma<9, true, true><<<dim3(32, 4, 4), 256, 0, stream>>>(h1, w2t, snet_b2, h2, 256, 128);
  wprep_kernel<<<9 * 128 * 32 / 256, 256, 0, stream>>>(snet_w3, w3t, 128, 32, 9);
  conv_wmma<9, false, false><<<dim3(32, 1, 4), 256, 0, stream>>>(h2, w3t, snet_b3, h3, 128, 32);
  gram_wmma_kernel<<<4, 32, 0, stream>>>(h3, gramB);
  fc_kernel<<<16, 256, 0, stream>>>(gramB, snet_fcw, snet_fcb, matS);

  trans_kernel<<<4, 1024, 0, stream>>>(matS, matC, trans_out);
  m2_kernel<<<256, 256, 0, stream>>>(uzw, trans_out, m2buf);
  final_kernel<<<32768, 256, 0, stream>>>(m2buf, ccflat, uzb, sMout, out);
}
